// SketchLinear_12721693131116
// MI455X (gfx1250) — matrix operand order repaired and checked by reading, not compiled, not run on hardware
//
#include <hip/hip_runtime.h>

// Problem dims (fixed by the reference).
#define BDIM 4096   // batch rows of input
#define DDIM 4096   // input features (sketched dimension)
#define ODIM 4096   // output features (weight rows)
#define MBUCK 2048  // countsketch buckets == GEMM K

typedef _Float16 v16h __attribute__((ext_vector_type(16)));
typedef _Float16 v8h  __attribute__((ext_vector_type(8)));
typedef float    v8f  __attribute__((ext_vector_type(8)));

// All per-lane offsets in this problem fit in 32 bits (sketch arrays 16 MB,
// output 64 MB), so we index as (uniform SGPR base) + (uint byte offset) to
// get GVS-mode (saddr) global_load/store instead of per-lane 64-bit address
// materialization.
__device__ __forceinline__ const v8h* off8h(const _Float16* base, unsigned elt_off) {
    return (const v8h*)((const char*)base + (size_t)(elt_off * 2u));
}

// ---------------------------------------------------------------------------
// Phase 1: countsketch.  One wave per row (8 rows / 256-thread WG).
// Buckets live in LDS (8 * 2048 * 4B = 64 KB).  Random hash -> ds_add_f32
// atomics with few bank conflicts.  Output stored as split f16 (hi + lo
// residual) so phase 2 can feed v_wmma_f32_16x16x32_f16 directly while
// retaining ~22 mantissa bits (fp32-class accuracy after 3-term WMMA).
// Rows 0..4095 -> input sketch, rows 4096..8191 -> weight sketch.
// ---------------------------------------------------------------------------
__global__ __launch_bounds__(256) void sketch_kernel(
    const float* __restrict__ input, const float* __restrict__ weight,
    const int* __restrict__ hash_idx, const float* __restrict__ rand_sgn,
    _Float16* __restrict__ is_hi, _Float16* __restrict__ is_lo,
    _Float16* __restrict__ ws_hi, _Float16* __restrict__ ws_lo)
{
    extern __shared__ float lds[];           // 8 * MBUCK floats
    const int lane = threadIdx.x & 31;
    const int wave = threadIdx.x >> 5;
    const int row  = blockIdx.x * 8 + wave;  // 0..8191

    float* buck = lds + wave * MBUCK;        // this wave's private bucket slice
    for (int i = lane; i < MBUCK; i += 32) buck[i] = 0.0f;
    // No barrier needed: each wave exclusively owns its LDS slice and DS ops
    // are in-order within a wave.

    const float* src = (row < BDIM) ? input : weight;
    const unsigned srow = (row < BDIM) ? (unsigned)row : (unsigned)(row - BDIM);
    const unsigned sbase = srow * (unsigned)DDIM;

    // 4096 columns / (32 lanes * 4) = 32 iterations of float4 loads.
    for (unsigned d0 = (unsigned)lane * 4u; d0 < DDIM; d0 += 32u * 4u) {
        const float4 x = *(const float4*)(src + (sbase + d0));
        const int4   h = *(const int4*)(hash_idx + d0);
        const float4 s = *(const float4*)(rand_sgn + d0);
        atomicAdd(&buck[h.x], x.x * s.x);
        atomicAdd(&buck[h.y], x.y * s.y);
        atomicAdd(&buck[h.z], x.z * s.z);
        atomicAdd(&buck[h.w], x.w * s.w);
    }

    _Float16* hi = (row < BDIM) ? is_hi : ws_hi;
    _Float16* lo = (row < BDIM) ? is_lo : ws_lo;
    const unsigned obase = srow * (unsigned)MBUCK;
    for (unsigned j = (unsigned)lane; j < MBUCK; j += 32u) {
        const float v = buck[j];
        const _Float16 h = (_Float16)v;              // RTNE high part
        const _Float16 l = (_Float16)(v - (float)h); // exact residual, rounded
        hi[obase + j] = h;
        lo[obase + j] = l;
    }
}

// ---------------------------------------------------------------------------
// Phase 2: out = IS @ WS^T + bias via v_wmma_f32_16x16x32_f16.
//
// A-fragment layout (ISA 7.12.2, 16-bit A 16x32): lane L<16 -> row m0+L,
// K = {k..k+7} in V0-3 and {k+16..k+23} in V4-7; lane L>=16 -> same row,
// K = {k+8..k+15, k+24..k+31}.  Two contiguous b128 loads per lane.
// ---------------------------------------------------------------------------
__device__ __forceinline__ v16h load_a_frag(const _Float16* __restrict__ base,
                                            unsigned row, unsigned k, unsigned lane)
{
    const unsigned r  = row + (lane & 15u);
    const unsigned kk = k + ((lane & 16u) ? 8u : 0u);
    const unsigned e  = r * (unsigned)MBUCK + kk;
    const v8h q0 = *off8h(base, e);          // K kk..kk+7
    const v8h q1 = *off8h(base, e + 16u);    // K kk+16..kk+23
    v16h out;
#pragma unroll
    for (int i = 0; i < 8; ++i) { out[i] = q0[i]; out[i + 8] = q1[i]; }
    return out;
}

// B-fragment (32x16 K-major, B = WS^T so B[k][n] = WS[n][k]): lane n<16 holds
// column n with K = {k..k+7, k+16..k+23}; lane n+16 holds K={k+8..k+15, k+24..k+31}
// (same K order as the A fragment).  Two b128 loads from the row-major WS sketch.
__device__ __forceinline__ v16h load_b_frag(const _Float16* __restrict__ base,
                                            unsigned row, unsigned k, unsigned lane)
{
    const unsigned r  = row + (lane & 15u);
    const unsigned kk = k + ((lane & 16u) ? 8u : 0u);
    const unsigned e  = r * (unsigned)MBUCK + kk;
    const v8h q0 = *off8h(base, e);          // K kk..kk+7
    const v8h q1 = *off8h(base, e + 16u);    // K kk+16..kk+23
    v16h out;
#pragma unroll
    for (int i = 0; i < 8; ++i) { out[i] = q0[i]; out[i + 8] = q1[i]; }
    return out;
}

__global__ __launch_bounds__(256) void gemm_kernel(
    const _Float16* __restrict__ is_hi, const _Float16* __restrict__ is_lo,
    const _Float16* __restrict__ ws_hi, const _Float16* __restrict__ ws_lo,
    const float* __restrict__ bias, float* __restrict__ out)
{
    const unsigned lane = threadIdx.x & 31u;
    const unsigned wave = threadIdx.x >> 5;
    const unsigned wm = wave >> 2;                 // 0..1
    const unsigned wn = wave & 3u;                 // 0..3
    const unsigned m0 = blockIdx.y * 128u + wm * 64u;
    const unsigned n0 = blockIdx.x * 128u + wn * 32u;

    v8f acc[4][2] = {};                       // 4 M-tiles x 2 N-tiles of 16x16

    for (unsigned k = 0; k < MBUCK; k += 32u) {
        // Hoist ALL 12 fragments (24 b128 loads) before the compute loops so
        // they live simultaneously: the scheduler clauses the loads and drains
        // them with staggered s_wait_loadcnt against the 24 WMMAs, instead of
        // serializing load->wait0->wmma per m-tile.
        v16h bh[2], bl[2], ah[4], al[4];
#pragma unroll
        for (unsigned j = 0; j < 2; ++j) {
            bh[j] = load_b_frag(ws_hi, n0 + 16u * j, k, lane);
            bl[j] = load_b_frag(ws_lo, n0 + 16u * j, k, lane);
        }
#pragma unroll
        for (unsigned i = 0; i < 4; ++i) {
            ah[i] = load_a_frag(is_hi, m0 + 16u * i, k, lane);
            al[i] = load_a_frag(is_lo, m0 + 16u * i, k, lane);
        }
#pragma unroll
        for (unsigned i = 0; i < 4; ++i) {
#pragma unroll
            for (unsigned j = 0; j < 2; ++j) {
                // fp32-class product: hi*hi + hi*lo + lo*hi (lo*lo ~ 2^-22, dropped)
                acc[i][j] = __builtin_amdgcn_wmma_f32_16x16x32_f16(
                    false, ah[i], false, bh[j], (short)0, acc[i][j], false, false);
                acc[i][j] = __builtin_amdgcn_wmma_f32_16x16x32_f16(
                    false, ah[i], false, bl[j], (short)0, acc[i][j], false, false);
                acc[i][j] = __builtin_amdgcn_wmma_f32_16x16x32_f16(
                    false, al[i], false, bh[j], (short)0, acc[i][j], false, false);
            }
        }
    }

    // Epilogue: C/D layout (32-bit 16x16): VGPR v, lane L<16 -> row v, col L;
    // lane L>=16 -> row v+8, col L-16.  Add bias[col], store fp32.
    // uint element offsets (max 16M < 2^32) keep stores in saddr form too.
    const unsigned col  = lane & 15u;
    const unsigned rsel = (lane & 16u) ? 8u : 0u;
#pragma unroll
    for (unsigned i = 0; i < 4; ++i) {
#pragma unroll
        for (unsigned j = 0; j < 2; ++j) {
            const unsigned n  = n0 + 16u * j + col;
            const float bv = bias[n];
            const unsigned ebase = (m0 + 16u * i + rsel) * (unsigned)ODIM + n;
#pragma unroll
            for (unsigned v = 0; v < 8; ++v) {
                out[ebase + v * (unsigned)ODIM] = acc[i][j][v] + bv;
            }
        }
    }
}

// ---------------------------------------------------------------------------
// Launch: phase 1 fills the f16 hi/lo sketches in d_ws (4 * 16 MB = 64 MB),
// phase 2 consumes them (L2-resident: 64 MB < 192 MB L2) and writes d_out.
// ---------------------------------------------------------------------------
extern "C" void kernel_launch(void* const* d_in, const int* in_sizes, int n_in,
                              void* d_out, int out_size, void* d_ws, size_t ws_size,
                              hipStream_t stream)
{
    const float* input    = (const float*)d_in[0];
    const float* weight   = (const float*)d_in[1];
    const float* bias     = (const float*)d_in[2];
    const int*   hash_idx = (const int*)d_in[3];
    const float* rand_sgn = (const float*)d_in[4];
    float* out = (float*)d_out;

    _Float16* is_hi = (_Float16*)d_ws;
    _Float16* is_lo = is_hi + (size_t)BDIM * MBUCK;
    _Float16* ws_hi = is_lo + (size_t)BDIM * MBUCK;
    _Float16* ws_lo = ws_hi + (size_t)ODIM * MBUCK;

    // Phase 1: 8192 rows total, 8 rows per 256-thread WG, 64 KB dynamic LDS.
    const int sketch_blocks = (BDIM + ODIM) / 8;
    sketch_kernel<<<sketch_blocks, 256, 8 * MBUCK * sizeof(float), stream>>>(
        input, weight, hash_idx, rand_sgn, is_hi, is_lo, ws_hi, ws_lo);

    // Phase 2: 128x128 block tiles -> 32x32 grid of 256-thread WGs.
    dim3 ggrid(ODIM / 128, BDIM / 128);
    gemm_kernel<<<ggrid, 256, 0, stream>>>(is_hi, is_lo, ws_hi, ws_lo, bias, out);

    (void)in_sizes; (void)n_in; (void)out_size; (void)ws_size;
}
